// LSTMCharTagger_61572651155620
// MI455X (gfx1250) — compile-verified, matrix operand order checked by reading, not run on hardware
//
#include <hip/hip_runtime.h>
#include <hip/hip_bf16.h>

// ---------------------------------------------------------------------------
// LSTMCharTagger for MI455X (gfx1250, wave32, WMMA, 320KB LDS/WGP).
//
// v3:
//  - Char scan: 4 waves/block share LDS-resident Whh (128KB); the 4 gate
//    tiles (i,f,g,o) are computed together so the LSTM nonlinearity runs
//    entirely in registers in WMMA C-layout (no gate LDS round-trip).
//  - Native transcendentals (v_exp_f32/v_rcp_f32) for sigmoid/tanh: no
//    libm exec-mask branching in the gate phase; TRANS ops co-execute
//    with WMMA on CDNA5.
//  - Blocked 64x64 WMMA GEMMs for all input projections; word scan is a
//    persistent block per direction with global_prefetch on future Z rows.
// ---------------------------------------------------------------------------

typedef __attribute__((ext_vector_type(16))) __bf16 v16bf;
typedef __attribute__((ext_vector_type(8)))  float  v8f;

#define S_WORDS 8192
#define T_CH    16
#define TS      (T_CH * S_WORDS)   // 131072 char-step rows

union BF16Frag { v16bf v; __bf16 e[16]; };

// 16-bit A/B fragment layout (ISA 7.12.2): lane L -> row/col = L%16; half idx ->
// K within chunk; lanes>=16 shifted +8 per half. Per lane this is two contiguous
// 8-half (16B) runs, which the compiler merges into b128/ds_b128 loads.
__device__ __forceinline__ int kmap(int idx, int lane) {
  int v = idx >> 1, p = idx & 1;
  int hi = (lane >= 16) ? 8 : 0;
  return (v < 4) ? (hi + 2 * v + p) : (16 + hi + 2 * (v - 4) + p);
}

// Fast transcendentals on the TRANS pipe (v_exp_f32, v_rcp_f32); both
// saturate correctly for large |x| (exp2 -> inf -> rcp -> 0).
__device__ __forceinline__ float fexp(float x)  { return __builtin_amdgcn_exp2f(x * 1.44269504f); }
__device__ __forceinline__ float frcp(float x)  { return __builtin_amdgcn_rcpf(x); }
__device__ __forceinline__ float sigm(float x)  { return frcp(1.0f + fexp(-x)); }
__device__ __forceinline__ float ftanh(float x) { return 1.0f - 2.0f * frcp(fexp(2.0f * x) + 1.0f); }

// ---------------------------------------------------------------------------
// Weight conversion fp32->bf16 with optional zero padding.
// ---------------------------------------------------------------------------
__global__ void pack_weights_kernel(const float* __restrict__ ih, const float* __restrict__ hh,
                                    __bf16* __restrict__ out, int rows, int E, int H, int Kc,
                                    int ndir) {
  long long idx = (long long)blockIdx.x * blockDim.x + threadIdx.x;
  long long total = (long long)ndir * rows * Kc;
  if (idx >= total) return;
  int k = (int)(idx % Kc);
  long long rr = idx / Kc;
  int r = (int)(rr % rows);
  int dir = (int)(rr / rows);
  float v = 0.0f;
  if (k < E)          v = ih[((long long)dir * rows + r) * E + k];
  else if (k < E + H) v = hh[((long long)dir * rows + r) * H + (k - E)];
  out[idx] = (__bf16)v;
}

// ---------------------------------------------------------------------------
// Char embedding gather: xc[t*S+s][e] = char_emb[char_ix[s][t]][e]  (bf16)
// ---------------------------------------------------------------------------
__global__ void build_xc_kernel(const int* __restrict__ char_ix, const float* __restrict__ char_emb,
                                __bf16* __restrict__ xc) {
  long long idx = (long long)blockIdx.x * blockDim.x + threadIdx.x;
  long long total = (long long)TS * 64;
  if (idx >= total) return;
  int e = (int)(idx % 64);
  long long r = idx / 64;                 // r = t*S + s
  int t = (int)(r / S_WORDS);
  int s = (int)(r % S_WORDS);
  int ci = char_ix[s * T_CH + t];
  xc[idx] = (__bf16)char_emb[(long long)ci * 64 + e];
}

// ---------------------------------------------------------------------------
// Word input build: xw[s][k] (bf16, 556 padded to 576)
// ---------------------------------------------------------------------------
__global__ void build_xw_kernel(const int* __restrict__ word_ix, const float* __restrict__ word_emb,
                                const float* __restrict__ crep, __bf16* __restrict__ xw) {
  long long idx = (long long)blockIdx.x * blockDim.x + threadIdx.x;
  long long total = (long long)S_WORDS * 576;
  if (idx >= total) return;
  int k = (int)(idx % 576);
  int s = (int)(idx / 576);
  float v = 0.0f;
  if (k < 300)      v = word_emb[(long long)word_ix[s] * 300 + k];
  else if (k < 556) v = crep[(long long)s * 256 + (k - 300)];
  xw[idx] = (__bf16)v;
}

// ---------------------------------------------------------------------------
// Blocked WMMA GEMM: Z[dir][m][n] = sum_k A[m][k] * W[dir][n][k]
// Block = 128 threads (4 waves), 64x64 macro-tile, K staged in 32-chunks via LDS.
// ---------------------------------------------------------------------------
template <bool OUT_BF16>
__global__ void __launch_bounds__(128)
gemm_bf16_kernel(const __bf16* __restrict__ A, const __bf16* __restrict__ W,
                 void* __restrict__ Zout, int M, int N, int K,
                 long long w_dir_stride, long long z_dir_stride) {
  __shared__ __bf16 As[64 * 32];   // 4KB
  __shared__ __bf16 Ws[64 * 32];   // 4KB
  const int tid = threadIdx.x;
  const int lane = tid & 31;
  const int wv = tid >> 5;                  // wave id 0..3 -> m sub-tile
  const int ntiles = N / 64;
  const int m0 = (blockIdx.x / ntiles) * 64;
  const int n0 = (blockIdx.x % ntiles) * 64;
  const int d = blockIdx.y;
  const int row = lane & 15;

  const __bf16* Wd = W + w_dir_stride * d;

  v8f acc[4];
  #pragma unroll
  for (int ns = 0; ns < 4; ++ns) acc[ns] = v8f{0.f, 0.f, 0.f, 0.f, 0.f, 0.f, 0.f, 0.f};

  const int srow = tid >> 1;                // 0..63
  const int sseg = (tid & 1) * 16;          // half-row segment (16 halves = 32B)

  for (int k0 = 0; k0 < K; k0 += 32) {
    {
      const uint4* ag = (const uint4*)(A + (long long)(m0 + srow) * K + k0 + sseg);
      const uint4* wg = (const uint4*)(Wd + (long long)(n0 + srow) * K + k0 + sseg);
      uint4* as = (uint4*)&As[srow * 32 + sseg];
      uint4* ws = (uint4*)&Ws[srow * 32 + sseg];
      as[0] = ag[0]; as[1] = ag[1];
      ws[0] = wg[0]; ws[1] = wg[1];
    }
    __syncthreads();
    BF16Frag Af;
    #pragma unroll
    for (int idx = 0; idx < 16; ++idx)
      Af.e[idx] = As[(wv * 16 + row) * 32 + kmap(idx, lane)];
    #pragma unroll
    for (int ns = 0; ns < 4; ++ns) {
      BF16Frag Bf;
      #pragma unroll
      for (int idx = 0; idx < 16; ++idx)
        Bf.e[idx] = Ws[(ns * 16 + row) * 32 + kmap(idx, lane)];
      acc[ns] = __builtin_amdgcn_wmma_f32_16x16x32_bf16(false, Af.v, false, Bf.v,
                                                        (short)0, acc[ns], false, false);
    }
    __syncthreads();
  }

  const int mbase = (lane >= 16) ? 8 : 0;
  #pragma unroll
  for (int ns = 0; ns < 4; ++ns) {
    #pragma unroll
    for (int j = 0; j < 8; ++j) {
      long long off = z_dir_stride * d +
                      (long long)(m0 + wv * 16 + j + mbase) * N + (n0 + ns * 16 + row);
      if (OUT_BF16) ((__bf16*)Zout)[off] = (__bf16)acc[ns][j];
      else          ((float*)Zout)[off]  = acc[ns][j];
    }
  }
}

// ---------------------------------------------------------------------------
// Char LSTM recurrent scan. 4 waves/block, 64 batch rows/block, 16 timesteps.
// Whh (128KB bf16) staged once into LDS, shared by all 4 waves for all steps.
// Gate tiles i/f/g/o (columns k, k+128, k+256, k+384) are computed together:
// for fixed (m,k) all four live at the same lane/slot of the 4 accumulators,
// so the LSTM cell update happens fully in registers (C-layout); only h goes
// back to LDS (bf16) as next step's A-matrix source.
// Both directions scan FORWARD (matches the reference's NOTE).
//   XP  : bf16 [2][T*S][512]  (x_t @ Wih^T, precomputed)
//   Whh : bf16 [2][512][128]
//   Oout: bf16 [T*S][256]  (layer0; null for layer1)
//   crep: f32  [S][256]    (layer1 only, gathered at t=char_len-1)
// ---------------------------------------------------------------------------
__global__ void __launch_bounds__(128)
char_scan_kernel(const __bf16* __restrict__ XP,
                 const __bf16* __restrict__ Whh,
                 const float* __restrict__ bias,   // [2][512]
                 __bf16* __restrict__ Oout,
                 float* __restrict__ crep,
                 const int* __restrict__ char_len) {
  const int tid = threadIdx.x;
  const int lane = tid & 31;
  const int wv = tid >> 5;                 // wave id 0..3
  const int r0 = blockIdx.x * 64;          // 64 batch rows per block
  const int d  = blockIdx.y;

  __shared__ __bf16 Ws[512 * 128];   // 128KB: Whh[d], resident all 16 steps
  __shared__ __bf16 hb[64 * 128];    // 16KB: h state per wave slice

  // stage Whh[d] into LDS (8192 uint4 across 128 threads)
  {
    const uint4* src = (const uint4*)(Whh + (long long)d * 512 * 128);
    uint4* dst = (uint4*)Ws;
    for (int i = tid; i < 512 * 128 / 8; i += 128) dst[i] = src[i];
  }
  // init h slice of this wave
  for (int m = 0; m < 16; ++m)
    for (int ks = 0; ks < 4; ++ks)
      hb[(wv * 16 + m) * 128 + (lane + 32 * ks)] = (__bf16)0.0f;

  float creg[64];                          // c state: [nt 0..7][j 0..7] in C-layout
  #pragma unroll
  for (int i = 0; i < 64; ++i) creg[i] = 0.0f;

  // hoist bias: depends only on column k = nt*16 + (lane&15)
  const int col = lane & 15;
  float bI[8], bF[8], bG[8], bO[8];
  #pragma unroll
  for (int nt = 0; nt < 8; ++nt) {
    int k = nt * 16 + col;
    bI[nt] = bias[d * 512 + k];
    bF[nt] = bias[d * 512 + 128 + k];
    bG[nt] = bias[d * 512 + 256 + k];
    bO[nt] = bias[d * 512 + 384 + k];
  }
  __syncthreads();

  const int mrow = lane & 15;
  const int mbase = (lane >= 16) ? 8 : 0;
  const long long hoff = (long long)(wv * 16) * 128;

  for (int t = 0; t < T_CH; ++t) {
    // A fragments of h (K=128 -> 4 chunks), reused across all gate tiles
    BF16Frag A[4];
    #pragma unroll
    for (int ch = 0; ch < 4; ++ch) {
      #pragma unroll
      for (int idx = 0; idx < 16; ++idx)
        A[ch].e[idx] = hb[hoff + mrow * 128 + ch * 32 + kmap(idx, lane)];
    }
    #pragma unroll 1
    for (int nt = 0; nt < 8; ++nt) {       // hidden-column tile: k in [nt*16, nt*16+16)
      v8f gi = {0.f,0.f,0.f,0.f,0.f,0.f,0.f,0.f};
      v8f gf = gi, gg = gi, go = gi;
      #pragma unroll
      for (int ch = 0; ch < 4; ++ch) {
        BF16Frag Bi, Bf_, Bg, Bo;
        #pragma unroll
        for (int idx = 0; idx < 16; ++idx) {
          int kk = ch * 32 + kmap(idx, lane);
          Bi.e[idx]  = Ws[((nt     ) * 16 + mrow) * 128 + kk];
          Bf_.e[idx] = Ws[((nt +  8) * 16 + mrow) * 128 + kk];
          Bg.e[idx]  = Ws[((nt + 16) * 16 + mrow) * 128 + kk];
          Bo.e[idx]  = Ws[((nt + 24) * 16 + mrow) * 128 + kk];
        }
        gi = __builtin_amdgcn_wmma_f32_16x16x32_bf16(false, A[ch].v, false, Bi.v,  (short)0, gi, false, false);
        gf = __builtin_amdgcn_wmma_f32_16x16x32_bf16(false, A[ch].v, false, Bf_.v, (short)0, gf, false, false);
        gg = __builtin_amdgcn_wmma_f32_16x16x32_bf16(false, A[ch].v, false, Bg.v,  (short)0, gg, false, false);
        go = __builtin_amdgcn_wmma_f32_16x16x32_bf16(false, A[ch].v, false, Bo.v,  (short)0, go, false, false);
      }
      // gate math in registers (C-layout): slot j -> row m = j + mbase, col k
      const int k = nt * 16 + col;
      #pragma unroll
      for (int j = 0; j < 8; ++j) {
        const int m = j + mbase;
        const int r = r0 + wv * 16 + m;
        const __bf16* xp = XP + ((long long)d * TS + (long long)t * S_WORDS + r) * 512;
        float iv = sigm (gi[j] + (float)xp[      k] + bI[nt]);
        float fv = sigm (gf[j] + (float)xp[128 + k] + bF[nt]);
        float gv = ftanh(gg[j] + (float)xp[256 + k] + bG[nt]);
        float ov = sigm (go[j] + (float)xp[384 + k] + bO[nt]);
        float cn = fv * creg[nt * 8 + j] + iv * gv;
        float hn = ov * ftanh(cn);
        creg[nt * 8 + j] = cn;
        hb[hoff + m * 128 + k] = (__bf16)hn;
        if (Oout)
          Oout[((long long)t * S_WORDS + r) * 256 + d * 128 + k] = (__bf16)hn;
        if (crep && (char_len[r] - 1 == t))
          crep[(long long)r * 256 + d * 128 + k] = hn;
      }
    }
    __syncthreads();
  }
}

// ---------------------------------------------------------------------------
// Word LSTM scan (batch=1, sequential over S). One persistent block per
// direction (grid=2), 1024 threads = one gate preactivation per thread.
// ---------------------------------------------------------------------------
__global__ void __launch_bounds__(1024)
word_lstm_kernel(const float* __restrict__ Z,      // [2][S][1024]
                 const float* __restrict__ Whh,    // [2][1024][256]
                 const float* __restrict__ bias,   // [2][1024]
                 __bf16* __restrict__ out) {       // [S][512]
  const int j = threadIdx.x;
  const int dir = blockIdx.x;
  __shared__ float hs[256];
  __shared__ float gb[1024];
  float creg = 0.0f;
  if (j < 256) hs[j] = 0.0f;
  __syncthreads();
  const float* wr = Whh + ((long long)dir * 1024 + j) * 256;
  const float bj = bias[dir * 1024 + j];
  for (int step = 0; step < S_WORDS; ++step) {
    const int t = dir ? (S_WORDS - 1 - step) : step;   // dir1 = reverse scan
    const int tpre = dir ? (t - 8) : (t + 8);
    if (tpre >= 0 && tpre < S_WORDS)
      __builtin_prefetch(&Z[((long long)dir * S_WORDS + tpre) * 1024 + j], 0, 1);
    float acc = Z[((long long)dir * S_WORDS + t) * 1024 + j] + bj;
    #pragma unroll 8
    for (int k = 0; k < 256; ++k) acc += hs[k] * wr[k];
    gb[j] = acc;
    __syncthreads();
    if (j < 256) {
      float iv = sigm(gb[j]);
      float fv = sigm(gb[256 + j]);
      float gv = ftanh(gb[512 + j]);
      float ov = sigm(gb[768 + j]);
      creg = fv * creg + iv * gv;
      float hn = ov * ftanh(creg);
      hs[j] = hn;
      out[(long long)t * 512 + dir * 256 + j] = (__bf16)hn;
    }
    __syncthreads();
  }
}

// ---------------------------------------------------------------------------
// Final: logits = wo1 @ out_W^T + b ; log_softmax over 50 tags. One block/row.
// (Standard expf/logf here for output accuracy; negligible cost.)
// ---------------------------------------------------------------------------
__global__ void __launch_bounds__(64)
tag_kernel(const __bf16* __restrict__ wo1, const float* __restrict__ outW,
           const float* __restrict__ outb, float* __restrict__ out) {
  const int s = blockIdx.x;
  const int tid = threadIdx.x;
  __shared__ float lg[50];
  if (tid < 50) {
    float acc = outb[tid];
    const float* w = outW + (long long)tid * 512;
    const __bf16* x = wo1 + (long long)s * 512;
    #pragma unroll 8
    for (int k = 0; k < 512; ++k) acc += (float)x[k] * w[k];
    lg[tid] = acc;
  }
  __syncthreads();
  if (tid < 50) {
    float mx = -1e30f;
    for (int i = 0; i < 50; ++i) mx = fmaxf(mx, lg[i]);
    float sum = 0.0f;
    for (int i = 0; i < 50; ++i) sum += expf(lg[i] - mx);
    out[(long long)s * 50 + tid] = lg[tid] - mx - logf(sum);
  }
}

// ---------------------------------------------------------------------------
extern "C" void kernel_launch(void* const* d_in, const int* in_sizes, int n_in,
                              void* d_out, int out_size, void* d_ws, size_t ws_size,
                              hipStream_t stream) {
  (void)in_sizes; (void)n_in; (void)out_size; (void)ws_size;
  const int*   word_ix  = (const int*)d_in[0];
  const int*   char_ix  = (const int*)d_in[1];
  const int*   char_len = (const int*)d_in[2];
  const float* word_emb = (const float*)d_in[3];
  const float* char_emb = (const float*)d_in[4];
  const float* cW_ih0 = (const float*)d_in[5];
  const float* cW_hh0 = (const float*)d_in[6];
  const float* cb0    = (const float*)d_in[7];
  const float* cW_ih1 = (const float*)d_in[8];
  const float* cW_hh1 = (const float*)d_in[9];
  const float* cb1    = (const float*)d_in[10];
  const float* wW_ih0 = (const float*)d_in[11];
  const float* wW_hh0 = (const float*)d_in[12];
  const float* wb0    = (const float*)d_in[13];
  const float* wW_ih1 = (const float*)d_in[14];
  const float* wW_hh1 = (const float*)d_in[15];
  const float* wb1    = (const float*)d_in[16];
  const float* out_W  = (const float*)d_in[17];
  const float* out_b  = (const float*)d_in[18];
  float* out = (float*)d_out;

  char* ws = (char*)d_ws;
  size_t off = 0;
  auto take = [&](size_t bytes) -> char* {
    char* p = ws + off;
    off = (off + bytes + 255) & ~(size_t)255;
    return p;
  };
  __bf16* CWI0 = (__bf16*)take((size_t)2 * 512 * 64 * 2);
  __bf16* CWH0 = (__bf16*)take((size_t)2 * 512 * 128 * 2);
  __bf16* CWI1 = (__bf16*)take((size_t)2 * 512 * 256 * 2);
  __bf16* CWH1 = (__bf16*)take((size_t)2 * 512 * 128 * 2);
  __bf16* WW0  = (__bf16*)take((size_t)2 * 1024 * 576 * 2);
  __bf16* WW1  = (__bf16*)take((size_t)2 * 1024 * 512 * 2);
  __bf16* XC   = (__bf16*)take((size_t)TS * 64 * 2);
  __bf16* XP   = (__bf16*)take((size_t)2 * TS * 512 * 2);   // reused: layer0 then layer1
  __bf16* O0   = (__bf16*)take((size_t)TS * 256 * 2);
  float*  CREP = (float*) take((size_t)S_WORDS * 256 * 4);
  __bf16* XW   = (__bf16*)take((size_t)S_WORDS * 576 * 2);
  float*  Z    = (float*) take((size_t)2 * S_WORDS * 1024 * 4); // reused: l0 then l1
  __bf16* WO0  = (__bf16*)take((size_t)S_WORDS * 512 * 2);
  __bf16* WO1  = (__bf16*)take((size_t)S_WORDS * 512 * 2);

  auto cdiv = [](long long a, long long b) { return (unsigned)((a + b - 1) / b); };

  // 1) weight conversions (bf16)
  pack_weights_kernel<<<cdiv(2LL*512*64,  256), 256, 0, stream>>>(cW_ih0, nullptr, CWI0, 512, 64,  0, 64,  2);
  pack_weights_kernel<<<cdiv(2LL*512*128, 256), 256, 0, stream>>>(cW_hh0, nullptr, CWH0, 512, 128, 0, 128, 2);
  pack_weights_kernel<<<cdiv(2LL*512*256, 256), 256, 0, stream>>>(cW_ih1, nullptr, CWI1, 512, 256, 0, 256, 2);
  pack_weights_kernel<<<cdiv(2LL*512*128, 256), 256, 0, stream>>>(cW_hh1, nullptr, CWH1, 512, 128, 0, 128, 2);
  pack_weights_kernel<<<cdiv(2LL*1024*576,256), 256, 0, stream>>>(wW_ih0, nullptr, WW0, 1024, 556, 0, 576, 2);
  pack_weights_kernel<<<cdiv(2LL*1024*512,256), 256, 0, stream>>>(wW_ih1, nullptr, WW1, 1024, 512, 0, 512, 2);

  // 2) char embedding gather -> XC [T*S][64]
  build_xc_kernel<<<cdiv((long long)TS * 64, 256), 256, 0, stream>>>(char_ix, char_emb, XC);

  // 3) layer0 input projections: XP[dir] = XC @ CWI0[dir]^T  (M=131072,N=512,K=64)
  gemm_bf16_kernel<true><<<dim3((TS/64) * (512/64), 2), 128, 0, stream>>>(
      XC, CWI0, XP, TS, 512, 64, (long long)512 * 64, (long long)TS * 512);

  // 4) char layer0 recurrent scan (Whh in LDS) -> O0
  char_scan_kernel<<<dim3(S_WORDS / 64, 2), 128, 0, stream>>>(XP, CWH0, cb0, O0, nullptr, nullptr);

  // 5) layer1 input projections: XP[dir] = O0 @ CWI1[dir]^T  (K=256)
  gemm_bf16_kernel<true><<<dim3((TS/64) * (512/64), 2), 128, 0, stream>>>(
      O0, CWI1, XP, TS, 512, 256, (long long)512 * 256, (long long)TS * 512);

  // 6) char layer1 recurrent scan -> chars_repr (fused ragged gather)
  char_scan_kernel<<<dim3(S_WORDS / 64, 2), 128, 0, stream>>>(XP, CWH1, cb1, nullptr, CREP, char_len);

  // 7) word input build
  build_xw_kernel<<<cdiv((long long)S_WORDS * 576, 256), 256, 0, stream>>>(word_ix, word_emb, CREP, XW);

  // 8) word layer0 projections: Z[dir] = XW @ WW0[dir]^T  (M=8192,N=1024,K=576)
  gemm_bf16_kernel<false><<<dim3((S_WORDS/64) * (1024/64), 2), 128, 0, stream>>>(
      XW, WW0, Z, S_WORDS, 1024, 576, (long long)1024 * 576, (long long)S_WORDS * 1024);

  // 9) word LSTM layer0 scan -> WO0
  word_lstm_kernel<<<2, 1024, 0, stream>>>(Z, wW_hh0, wb0, WO0);

  // 10) word layer1 projections: Z[dir] = WO0 @ WW1[dir]^T  (K=512)
  gemm_bf16_kernel<false><<<dim3((S_WORDS/64) * (1024/64), 2), 128, 0, stream>>>(
      WO0, WW1, Z, S_WORDS, 1024, 512, (long long)1024 * 512, (long long)S_WORDS * 1024);

  // 11) word LSTM layer1 scan -> WO1
  word_lstm_kernel<<<2, 1024, 0, stream>>>(Z, wW_hh1, wb1, WO1);

  // 12) projection + log_softmax
  tag_kernel<<<S_WORDS, 64, 0, stream>>>(WO1, out_W, out_b, out);
}
